// GroupedQueryAttention_40578851012925
// MI455X (gfx1250) — compile-verified, matrix operand order checked by reading, not run on hardware
//
#include <hip/hip_runtime.h>

// ---------------------------------------------------------------------------
// GQA attention for MI455X (gfx1250, wave32, WMMA + TDM).
// Pipeline: f32->bf16 convert -> WMMA GEMM (Q/K/V proj) -> RoPE -> flash
// attention (WMMA QK^T + online softmax + WMMA PV) -> WMMA GEMM (out proj).
// Tile staging to LDS uses the Tensor Data Mover where the toolchain exposes
// __builtin_amdgcn_tensor_load_to_lds, with a vector-load fallback.
// ---------------------------------------------------------------------------

#define Bc 2
#define Sc 2048
#define Ec 2048
#define Hc 16
#define KVHc 4
#define HDc 128

typedef __attribute__((ext_vector_type(16))) __bf16 bf16x16;
typedef __attribute__((ext_vector_type(8)))  float  f32x8;
typedef __attribute__((ext_vector_type(4)))  unsigned int u32x4;
typedef __attribute__((ext_vector_type(8)))  int i32x8;
typedef __attribute__((ext_vector_type(4)))  int i32x4;

struct alignas(16) B128 { unsigned int w[4]; };

#if defined(__AMDGCN__) && __has_builtin(__builtin_amdgcn_tensor_load_to_lds)
#define HAS_TDM 1
#else
#define HAS_TDM 0
#endif

// 32-bit LDS byte offset of a __shared__ object (for TDM D# lds_addr).
__device__ inline unsigned lds_off(const void* p) {
#if defined(__AMDGCN__)
  return (unsigned)(unsigned long long)(
      __attribute__((address_space(3))) const char*)p;
#else
  (void)p;
  return 0u;
#endif
}

__device__ inline void wait_tensor0() {
#if defined(__AMDGCN__) && __has_builtin(__builtin_amdgcn_s_wait_tensorcnt)
  __builtin_amdgcn_s_wait_tensorcnt((short)0);
#elif defined(__AMDGCN__)
  asm volatile("s_wait_tensorcnt 0x0" ::: "memory");
#endif
}

// TDM 2-D tile load: global (row-major, row_stride elems of 2B) -> LDS with
// optional per-row padding. D# per CDNA5 ISA ch.8: group0 = {count, lds_addr,
// global_addr[56:0], type=2}, group1 = {data_size=2B, pad cfg, tensor dims,
// tile dims, dim0 stride}; groups 2/3 zero (2-D tensor).
__device__ inline void tdm_load_2d(unsigned lds_byte_addr, const void* gaddr,
                                   unsigned tile_w, unsigned tile_h,
                                   unsigned row_stride,
                                   unsigned pad_interval_enc,
                                   unsigned pad_amount_enc) {
#if HAS_TDM
  unsigned long long ga = (unsigned long long)gaddr;
  u32x4 g0;
  g0[0] = 1u;                                   // count=1, user descriptor
  g0[1] = lds_byte_addr;                        // lds_addr [63:32]
  g0[2] = (unsigned)(ga & 0xFFFFFFFFull);       // global_addr [95:64]
  g0[3] = (unsigned)((ga >> 32) & 0x1FFFFFFull) // global_addr [120:96]
          | (2u << 30);                         // type=2 ("image")
  i32x8 g1;
  unsigned d0 = (1u << 16)                      // data_size = 2 bytes
              | (1u << 20)                      // pad_enable
              | (pad_interval_enc << 22) | (pad_amount_enc << 25);
  g1[0] = (int)d0;
  g1[1] = (int)((tile_w & 0xFFFFu) << 16);            // tensor_dim0 lo
  g1[2] = (int)((tile_w >> 16) | ((tile_h & 0xFFFFu) << 16)); // td0 hi|td1 lo
  g1[3] = (int)((tile_h >> 16) | ((tile_w & 0xFFFFu) << 16)); // td1 hi|tile_dim0
  g1[4] = (int)(tile_h & 0xFFFFu);                    // tile_dim1 (tile_dim2=0)
  g1[5] = (int)row_stride;                            // tensor_dim0_stride lo
  g1[6] = 0;
  g1[7] = 0;
  i32x4 z4 = {0, 0, 0, 0};
#if __clang_major__ >= 23
  i32x8 z8 = {0, 0, 0, 0, 0, 0, 0, 0};
  __builtin_amdgcn_tensor_load_to_lds(g0, g1, z4, z4, z8, 0);
#else
  __builtin_amdgcn_tensor_load_to_lds(g0, g1, z4, z4, 0);
#endif
#else
  (void)lds_byte_addr; (void)gaddr; (void)tile_w; (void)tile_h;
  (void)row_stride; (void)pad_interval_enc; (void)pad_amount_enc;
#endif
}

// Load one A/B fragment for v_wmma_f32_16x16x32_bf16 from an LDS row.
// 16-bit A-matrix layout (ISA 7.12.2): lane<16 holds K = 0..7 and 16..23,
// lane>=16 holds K = 8..15 and 24..31 -> two contiguous 16B chunks per lane.
__device__ inline bf16x16 load_frag(const __bf16* rowptr, int half) {
  union { bf16x16 v; B128 b[2]; } u;
  u.b[0] = *reinterpret_cast<const B128*>(rowptr + 8 * half);
  u.b[1] = *reinterpret_cast<const B128*>(rowptr + 16 + 8 * half);
  return u.v;
}

__device__ inline f32x8 wmma_bf16(bf16x16 a, bf16x16 b, f32x8 c) {
  return __builtin_amdgcn_wmma_f32_16x16x32_bf16(false, a, false, b,
                                                 (short)0, c, false, false);
}

// ---------------------------------------------------------------------------
// Elementwise f32 -> bf16
// ---------------------------------------------------------------------------
__global__ void cvt_f32_bf16(const float* __restrict__ in,
                             __bf16* __restrict__ out, int n) {
  int i = blockIdx.x * blockDim.x + threadIdx.x;
  if (i < n) out[i] = (__bf16)in[i];
}

// ---------------------------------------------------------------------------
// Tiled bf16 WMMA GEMM:  C[M,N] = A[M,K] * W[K,N]   (A,W row-major bf16)
// Block = 128 threads (4 waves). Block tile 128x64, K staged 32 at a time.
// Wave w computes rows [w*32, w*32+32) x 64 cols: 8 WMMAs per K-step with
// 2 A-fragments + 4 B-fragments. A tile comes in via TDM (wave 0 issues,
// s_wait_tensorcnt, barrier); W tile is transposed into LDS by all threads.
// ---------------------------------------------------------------------------
template <bool OUT_F32>
__global__ __launch_bounds__(128) void gemm_bf16_wmma(
    const __bf16* __restrict__ A, const __bf16* __restrict__ W,
    float* __restrict__ Cf, __bf16* __restrict__ Cb, int M, int N, int K) {
  __shared__ __align__(16) __bf16 As[128][40];  // 128 rows x 32 K (+8 pad)
  __shared__ __align__(16) __bf16 Wt[64][40];   // 64 N-rows x 32 K (transposed)

  const int t = threadIdx.x;
  const int wave = t >> 5, lane = t & 31;
  const int half = lane >> 4, ln = lane & 15;
  const int m0 = blockIdx.y * 128, n0 = blockIdx.x * 64;

  f32x8 acc[2][4] = {};

  for (int kk = 0; kk < K; kk += 32) {
#if HAS_TDM
    // A tile 128x32 via TDM: row = 32 elems = 16 dwords (enc 3), pad = 4
    // dwords (enc 3) -> LDS row stride 40 elems.
    if (wave == 0 && lane == 0)
      tdm_load_2d(lds_off(&As[0][0]), A + (size_t)m0 * K + kk,
                  /*tile_w=*/32, /*tile_h=*/128, /*row_stride=*/(unsigned)K,
                  /*interval 16dw*/3, /*pad 4dw*/3);
#else
#pragma unroll
    for (int i = 0; i < 4; ++i) {
      int id = t * 4 + i;              // 0..511
      int row = id >> 2, cg = id & 3;  // 128 rows x 4 groups of 8
      B128 src = *reinterpret_cast<const B128*>(
          A + (size_t)(m0 + row) * K + kk + cg * 8);
      *reinterpret_cast<B128*>(&As[row][cg * 8]) = src;
    }
#endif
    // W tile: 32x64 read row-major (coalesced), stored transposed -> Wt[n][k].
#pragma unroll
    for (int i = 0; i < 2; ++i) {
      int id = t * 2 + i;               // 0..255
      int krow = id >> 3, ng = id & 7;  // 32 K-rows x 8 N-groups
      union { B128 b; __bf16 h[8]; } u;
      u.b = *reinterpret_cast<const B128*>(
          W + (size_t)(kk + krow) * N + n0 + ng * 8);
#pragma unroll
      for (int e = 0; e < 8; ++e) Wt[ng * 8 + e][krow] = u.h[e];
    }
#if HAS_TDM
    if (wave == 0) wait_tensor0();
#endif
    __syncthreads();

    bf16x16 af0 = load_frag(&As[wave * 32 + ln][0], half);
    bf16x16 af1 = load_frag(&As[wave * 32 + 16 + ln][0], half);
#pragma unroll
    for (int j = 0; j < 4; ++j) {
      bf16x16 bfr = load_frag(&Wt[j * 16 + ln][0], half);
      acc[0][j] = wmma_bf16(af0, bfr, acc[0][j]);
      acc[1][j] = wmma_bf16(af1, bfr, acc[1][j]);
    }
    __syncthreads();
  }

  // C layout: VGPR r, lane l -> row = r + 8*(l>>4), col = l&15 (per subtile).
#pragma unroll
  for (int a = 0; a < 2; ++a)
#pragma unroll
    for (int j = 0; j < 4; ++j)
#pragma unroll
      for (int r = 0; r < 8; ++r) {
        int row = m0 + wave * 32 + a * 16 + r + 8 * half;
        int col = n0 + j * 16 + ln;
        if (OUT_F32)
          Cf[(size_t)row * N + col] = acc[a][j][r];
        else
          Cb[(size_t)row * N + col] = (__bf16)acc[a][j][r];
      }
}

// ---------------------------------------------------------------------------
// In-place RoPE on [B, S, heads, 128] bf16. One thread per rotation pair.
// ---------------------------------------------------------------------------
__global__ void rope_kernel(__bf16* buf, int heads, int total) {
  int idx = blockIdx.x * blockDim.x + threadIdx.x;
  if (idx >= total) return;
  int j = idx & 63;  // half of head_dim
  int tmp = idx >> 6;
  int hh = tmp % heads;
  tmp /= heads;
  int s = tmp % Sc;
  int bb = tmp / Sc;
  // inv_freq = 10000^(-j/64);  ln(10000) = 9.210340371976184
  float f = __expf(-(float)j * (9.210340371976184f / 64.0f));
  float ang = (float)s * f;
  float cv = cosf(ang), sv = sinf(ang);
  size_t base = (((size_t)bb * Sc + s) * heads + hh) * HDc;
  float x0 = (float)buf[base + j];
  float x1 = (float)buf[base + j + 64];
  buf[base + j]      = (__bf16)(x0 * cv - x1 * sv);
  buf[base + j + 64] = (__bf16)(x1 * cv + x0 * sv);
}

// ---------------------------------------------------------------------------
// Flash attention (causal, GQA). Block: 128 threads = 4 waves; 64 query rows
// per block (16 per wave). KV tiles of 64 staged in LDS (K via TDM). WMMA for
// QK^T and PV; online softmax with cross-lane reductions in each 16-lane half.
// Q: [B,S,H*HD] bf16 (post-RoPE); K,V: [B,S,KVH*HD] bf16; O: [B,S,H*HD] bf16.
// ---------------------------------------------------------------------------
__global__ __launch_bounds__(128) void flash_attn(
    const __bf16* __restrict__ Q, const __bf16* __restrict__ Kg,
    const __bf16* __restrict__ Vg, __bf16* __restrict__ O) {
  const int qt = blockIdx.x;
  const int h  = blockIdx.y;
  const int b  = blockIdx.z;
  const int kvh = h / (Hc / KVHc);
  const int t = threadIdx.x, wave = t >> 5, lane = t & 31;
  const int half = lane >> 4, ln = lane & 15;
  const int q0 = qt * 64;

  __shared__ __align__(16) __bf16 Kt[64][136];    // kv-row major (+Q staging)
  __shared__ __align__(16) __bf16 Vt[128][72];    // d-major (transposed V)
  __shared__ __align__(16) __bf16 Ps[4][16][72];  // per-wave P staging

  // ---- stage Q tile (64x128) through Kt, pull persistent A-fragments ----
  // Row = 128 elems = 64 dwords (enc 5), pad = 8 elems = 4 dwords (enc 3).
#if HAS_TDM
  if (wave == 0 && lane == 0)
    tdm_load_2d(lds_off(&Kt[0][0]),
                Q + ((size_t)(b * Sc + q0)) * Ec + h * HDc,
                /*tile_w=*/128, /*tile_h=*/64, /*row_stride=*/Ec, 5, 3);
  if (wave == 0) wait_tensor0();
#else
#pragma unroll
  for (int i = 0; i < 8; ++i) {
    int id = i * 128 + t;
    int row = id >> 4, dg = id & 15;
    B128 src = *reinterpret_cast<const B128*>(
        Q + ((size_t)(b * Sc + q0 + row)) * Ec + h * HDc + dg * 8);
    *reinterpret_cast<B128*>(&Kt[row][dg * 8]) = src;
  }
#endif
  __syncthreads();
  bf16x16 qf[4];
#pragma unroll
  for (int c = 0; c < 4; ++c)
    qf[c] = load_frag(&Kt[wave * 16 + ln][32 * c], half);
  __syncthreads();

  f32x8 oacc[8] = {};
  float mrow[8], lrow[8];
#pragma unroll
  for (int r = 0; r < 8; ++r) { mrow[r] = -3.0e38f; lrow[r] = 0.0f; }

  const float scale = 0.08838834764831845f;  // 1/sqrt(128)
  const int ntiles = qt + 1;                 // causal: kv tiles <= diagonal

  for (int it = 0; it < ntiles; ++it) {
    const int kv0 = it * 64;
    // ---- K tile: 64 kv rows x 128 d, row-major; TDM from wave 0 ----
#if HAS_TDM
    if (wave == 0 && lane == 0)
      tdm_load_2d(lds_off(&Kt[0][0]),
                  Kg + ((size_t)(b * Sc + kv0)) * (KVHc * HDc) + kvh * HDc,
                  /*tile_w=*/128, /*tile_h=*/64, /*row_stride=*/KVHc * HDc,
                  5, 3);
#else
#pragma unroll
    for (int i = 0; i < 8; ++i) {
      int id = i * 128 + t;
      int row = id >> 4, dg = id & 15;
      B128 src = *reinterpret_cast<const B128*>(
          Kg + ((size_t)(b * Sc + kv0 + row)) * (KVHc * HDc) + kvh * HDc +
          dg * 8);
      *reinterpret_cast<B128*>(&Kt[row][dg * 8]) = src;
    }
#endif
    // ---- V tile: read row-major, store transposed Vt[d][kv] ----
#pragma unroll
    for (int i = 0; i < 8; ++i) {
      int id = i * 128 + t;
      int row = id >> 4, dg = id & 15;
      union { B128 b; __bf16 h[8]; } u;
      u.b = *reinterpret_cast<const B128*>(
          Vg + ((size_t)(b * Sc + kv0 + row)) * (KVHc * HDc) + kvh * HDc +
          dg * 8);
#pragma unroll
      for (int e = 0; e < 8; ++e) Vt[dg * 8 + e][row] = u.h[e];
      // Prefetch next iteration's V tile (-> global_prefetch_b8).
      if (it + 1 < ntiles)
        __builtin_prefetch(
            Vg + ((size_t)(b * Sc + kv0 + 64 + row)) * (KVHc * HDc) +
                kvh * HDc + dg * 8,
            0, 0);
    }
#if HAS_TDM
    if (wave == 0) wait_tensor0();
#endif
    __syncthreads();

    // ---- S = Q K^T : 4 N-subtiles x 4 K-chunks of WMMA ----
    f32x8 sc[4] = {};
#pragma unroll
    for (int j = 0; j < 4; ++j)
#pragma unroll
      for (int c = 0; c < 4; ++c) {
        bf16x16 kf = load_frag(&Kt[j * 16 + ln][32 * c], half);
        sc[j] = wmma_bf16(qf[c], kf, sc[j]);
      }

    // ---- scale + causal mask + row max ----
    float mcur[8];
#pragma unroll
    for (int r = 0; r < 8; ++r) mcur[r] = -3.0e38f;
#pragma unroll
    for (int j = 0; j < 4; ++j) {
      int col = kv0 + j * 16 + ln;
#pragma unroll
      for (int r = 0; r < 8; ++r) {
        int row = q0 + wave * 16 + r + 8 * half;
        float s = sc[j][r] * scale;
        if (col > row) s = -3.0e38f;
        sc[j][r] = s;
        mcur[r] = fmaxf(mcur[r], s);
      }
    }
#pragma unroll
    for (int r = 0; r < 8; ++r) {
      float v = mcur[r];
      v = fmaxf(v, __shfl_xor(v, 1));
      v = fmaxf(v, __shfl_xor(v, 2));
      v = fmaxf(v, __shfl_xor(v, 4));
      v = fmaxf(v, __shfl_xor(v, 8));
      mcur[r] = v;
    }
    // ---- online softmax rescale ----
#pragma unroll
    for (int r = 0; r < 8; ++r) {
      float mnew = fmaxf(mrow[r], mcur[r]);
      float alpha = __expf(mrow[r] - mnew);
      mrow[r] = mnew;
      lrow[r] *= alpha;
#pragma unroll
      for (int jd = 0; jd < 8; ++jd) oacc[jd][r] *= alpha;
    }
    // ---- P = exp(S-m) -> bf16 LDS staging + row sums ----
    float rsum[8];
#pragma unroll
    for (int r = 0; r < 8; ++r) rsum[r] = 0.0f;
#pragma unroll
    for (int j = 0; j < 4; ++j)
#pragma unroll
      for (int r = 0; r < 8; ++r) {
        float p = __expf(sc[j][r] - mrow[r]);
        rsum[r] += p;
        Ps[wave][r + 8 * half][j * 16 + ln] = (__bf16)p;
      }
#pragma unroll
    for (int r = 0; r < 8; ++r) {
      float v = rsum[r];
      v += __shfl_xor(v, 1);
      v += __shfl_xor(v, 2);
      v += __shfl_xor(v, 4);
      v += __shfl_xor(v, 8);
      lrow[r] += v;
    }

    // ---- O += P V : 2 K-chunks (kv) x 8 N-subtiles (d) ----
#pragma unroll
    for (int c2 = 0; c2 < 2; ++c2) {
      bf16x16 pf = load_frag(&Ps[wave][ln][32 * c2], half);
#pragma unroll
      for (int jd = 0; jd < 8; ++jd) {
        bf16x16 vf = load_frag(&Vt[jd * 16 + ln][32 * c2], half);
        oacc[jd] = wmma_bf16(pf, vf, oacc[jd]);
      }
    }
    __syncthreads();
  }

  // ---- normalize and write [B,S,H*HD] bf16 ----
#pragma unroll
  for (int jd = 0; jd < 8; ++jd)
#pragma unroll
    for (int r = 0; r < 8; ++r) {
      int row = q0 + wave * 16 + r + 8 * half;
      int col = h * HDc + jd * 16 + ln;
      O[((size_t)(b * Sc + row)) * Ec + col] = (__bf16)(oacc[jd][r] / lrow[r]);
    }
}

// ---------------------------------------------------------------------------
// Launcher
// ---------------------------------------------------------------------------
extern "C" void kernel_launch(void* const* d_in, const int* in_sizes, int n_in,
                              void* d_out, int out_size, void* d_ws,
                              size_t ws_size, hipStream_t stream) {
  (void)in_sizes; (void)n_in; (void)out_size; (void)ws_size;
  const float* x  = (const float*)d_in[0];
  const float* wq = (const float*)d_in[1];
  const float* wk = (const float*)d_in[2];
  const float* wv = (const float*)d_in[3];
  const float* wo = (const float*)d_in[4];
  float* out = (float*)d_out;

  const size_t M  = (size_t)Bc * Sc;     // 4096
  const size_t NE = Ec;                  // 2048
  const size_t NK = (size_t)KVHc * HDc;  // 512

  char* ws = (char*)d_ws;
  size_t off = 0;
  auto alloc = [&](size_t bytes) -> char* {
    off = (off + 255) & ~(size_t)255;
    char* p = ws + off;
    off += bytes;
    return p;
  };
  __bf16* xb  = (__bf16*)alloc(M * NE * 2);
  __bf16* wqb = (__bf16*)alloc(NE * NE * 2);
  __bf16* wkb = (__bf16*)alloc(NE * NK * 2);
  __bf16* wvb = (__bf16*)alloc(NE * NK * 2);
  __bf16* wob = (__bf16*)alloc(NE * NE * 2);
  __bf16* qb  = (__bf16*)alloc(M * NE * 2);
  __bf16* kb  = (__bf16*)alloc(M * NK * 2);
  __bf16* vb  = (__bf16*)alloc(M * NK * 2);
  __bf16* ab  = (__bf16*)alloc(M * NE * 2);

  auto cvt = [&](const float* src, __bf16* dst, size_t n) {
    cvt_f32_bf16<<<dim3((unsigned)((n + 255) / 256)), dim3(256), 0, stream>>>(
        src, dst, (int)n);
  };
  cvt(x,  xb,  M * NE);
  cvt(wq, wqb, NE * NE);
  cvt(wk, wkb, NE * NK);
  cvt(wv, wvb, NE * NK);
  cvt(wo, wob, NE * NE);

  // Q/K/V projections (block tile 128x64)
  gemm_bf16_wmma<false><<<dim3((unsigned)(NE / 64), (unsigned)(M / 128)),
                          dim3(128), 0, stream>>>(
      xb, wqb, nullptr, qb, (int)M, (int)NE, (int)NE);
  gemm_bf16_wmma<false><<<dim3((unsigned)(NK / 64), (unsigned)(M / 128)),
                          dim3(128), 0, stream>>>(
      xb, wkb, nullptr, kb, (int)M, (int)NK, (int)NE);
  gemm_bf16_wmma<false><<<dim3((unsigned)(NK / 64), (unsigned)(M / 128)),
                          dim3(128), 0, stream>>>(
      xb, wvb, nullptr, vb, (int)M, (int)NK, (int)NE);

  // RoPE (in place)
  {
    int totq = Bc * Sc * Hc * (HDc / 2);
    int totk = Bc * Sc * KVHc * (HDc / 2);
    rope_kernel<<<dim3((unsigned)((totq + 255) / 256)), dim3(256), 0, stream>>>(
        qb, Hc, totq);
    rope_kernel<<<dim3((unsigned)((totk + 255) / 256)), dim3(256), 0, stream>>>(
        kb, KVHc, totk);
  }

  // Flash attention
  flash_attn<<<dim3(Sc / 64, Hc, Bc), dim3(128), 0, stream>>>(qb, kb, vb, ab);

  // Output projection -> f32
  gemm_bf16_wmma<true><<<dim3((unsigned)(NE / 64), (unsigned)(M / 128)),
                         dim3(128), 0, stream>>>(
      ab, wob, out, nullptr, (int)M, (int)NE, (int)NE);
}
